// ElmoLSTM2_35416300323645
// MI455X (gfx1250) — matrix operand order repaired and checked by reading, not compiled, not run on hardware
//
#include <hip/hip_runtime.h>
#include <hip/hip_bf16.h>

// ---------------------------------------------------------------------------
// ELMo bidirectional 2-layer LSTM + vocab projection for MI455X (gfx1250).
// All GEMMs use v_wmma_f32_16x16x32_bf16 (bf16 operands, f32 accumulate).
// S=128, B=32, D=H=512, 4H=2048, V=8000.
// A-panels (32x512 bf16 = 32KB) staged in LDS once per block; barrier-free
// K-loop; each wave holds 4 accumulators (2 M-halves x 2 N columns) so each
// chunk is 4 ds_load_b128 + 4 global_load_b128 -> 4 wmma.
// ---------------------------------------------------------------------------

typedef unsigned short ushort_t;
typedef __attribute__((ext_vector_type(16))) __bf16        bf16x16;
typedef __attribute__((ext_vector_type(8)))  float         f32x8;
typedef __attribute__((ext_vector_type(8)))  unsigned int  u32x8;

#define S_LEN 128
#define BATCH 32
#define HID   512
#define GATES 2048    // 4*HID
#define VOCAB 8000
#define VPAD  8192    // VOCAB padded to multiple of 256
#define MROWS 4096    // S_LEN*BATCH

__device__ __forceinline__ ushort_t f32_to_bf16(float f) {
    unsigned u = __float_as_uint(f);
    u += 0x7FFFu + ((u >> 16) & 1u);          // round-to-nearest-even
    return (ushort_t)(u >> 16);
}
__device__ __forceinline__ float fast_sigmoid(float x) {
    return 1.0f / (1.0f + __expf(-x));
}
__device__ __forceinline__ float fast_tanh(float x) {
    float e = __expf(2.0f * x);               // x>>0 -> 1; x<<0 -> -1
    return 1.0f - 2.0f / (e + 1.0f);
}

// ---- fragment helpers ------------------------------------------------------
__device__ __forceinline__ bf16x16 pack_frag(uint4 lo, uint4 hi) {
    u32x8 u;
    u[0] = lo.x; u[1] = lo.y; u[2] = lo.z; u[3] = lo.w;
    u[4] = hi.x; u[5] = hi.y; u[6] = hi.z; u[7] = hi.w;
    return __builtin_bit_cast(bf16x16, u);
}
// A (16x32 bf16) from LDS panel [32][512]: lane row = rbase + (lane&15);
// regs 0..3 = K kb..kb+7, regs 4..7 = K 16+kb..16+kb+7, kb = (lane>>4)*8.
__device__ __forceinline__ bf16x16 load_a_frag(const ushort_t* lds_a, int rbase,
                                               int k0, int lane) {
    int m  = rbase + (lane & 15);
    int kb = (lane >> 4) * 8;
    const uint4* p = reinterpret_cast<const uint4*>(lds_a + m * HID + k0 + kb);
    return pack_frag(p[0], p[2]);             // +16 ushorts == +2 uint4
}
// B (32x16 bf16): W row-major [N,512]; regs 0..3 = K kb..kb+7, regs 4..7 = K 16+kb..16+kb+7, kb = (lane>>4)*8.
__device__ __forceinline__ bf16x16 load_b_frag(const ushort_t* __restrict__ W,
                                               int n, int k0, int lane) {
    int koff = (lane >> 4) * 8;
    const uint4* p = reinterpret_cast<const uint4*>(W + (size_t)n * HID + k0 + koff);
    return pack_frag(p[0], p[2]);
}

// Stage A rows m0..m0+31 (row-major, lda=512) into LDS: 32KB, 8 uint4/thread.
__device__ __forceinline__ void stage_a_panel(const ushort_t* __restrict__ A, int m0,
                                              ushort_t* lds_a) {
    const int t = threadIdx.x;
#pragma unroll
    for (int j = 0; j < 8; ++j) {
        int idx = t + j * 256;               // uint4 index, 2048 total
        int m   = idx >> 6;                  // 64 uint4 per 512-ushort row
        int kk  = (idx & 63) * 8;
        reinterpret_cast<uint4*>(lds_a + m * HID + kk)[0] =
            reinterpret_cast<const uint4*>(A + (size_t)(m0 + m) * HID + kk)[0];
    }
}

// Barrier-free K loop: 2 M-halves x 2 N columns per wave.
// Per chunk: 4 LDS b128 + 4 global b128 -> 4 wmma (each frag reused twice).
__device__ __forceinline__ void wave_gemm_quad(const ushort_t* lds_a,
                                               const ushort_t* __restrict__ W,
                                               int nA, int nB, int lane,
                                               f32x8& acc00, f32x8& acc10,
                                               f32x8& acc01, f32x8& acc11) {
#pragma unroll 2
    for (int k0 = 0; k0 < HID; k0 += 32) {
        bf16x16 a0 = load_a_frag(lds_a, 0,  k0, lane);
        bf16x16 a1 = load_a_frag(lds_a, 16, k0, lane);
        bf16x16 b0 = load_b_frag(W, nA, k0, lane);
        bf16x16 b1 = load_b_frag(W, nB, k0, lane);
        acc00 = __builtin_amdgcn_wmma_f32_16x16x32_bf16(false, a0, false, b0, (short)0, acc00,
                                                        false, false);
        acc10 = __builtin_amdgcn_wmma_f32_16x16x32_bf16(false, a1, false, b0, (short)0, acc10,
                                                        false, false);
        acc01 = __builtin_amdgcn_wmma_f32_16x16x32_bf16(false, a0, false, b1, (short)0, acc01,
                                                        false, false);
        acc11 = __builtin_amdgcn_wmma_f32_16x16x32_bf16(false, a1, false, b1, (short)0, acc11,
                                                        false, false);
    }
}

// ---- GEMM kernels ----------------------------------------------------------

// gates_in[M,N] = A[M,512] @ Wp[N,512]^T + biasp[N].  Block tile 32M x 256N.
__global__ void __launch_bounds__(256)
gates_in_gemm(const ushort_t* __restrict__ A, const ushort_t* __restrict__ Wp,
              const float* __restrict__ biasp, float* __restrict__ out, int N) {
    __shared__ __attribute__((aligned(16))) ushort_t lds_a[32 * HID];
    int tid = threadIdx.x, lane = tid & 31, wave = tid >> 5;
    int m0 = blockIdx.y * 32;
    int nA = blockIdx.x * 256 + wave * 32 + (lane & 15);
    int nB = nA + 16;
    stage_a_panel(A, m0, lds_a);
    __syncthreads();
    f32x8 acc00 = {}, acc10 = {}, acc01 = {}, acc11 = {};
    wave_gemm_quad(lds_a, Wp, nA, nB, lane, acc00, acc10, acc01, acc11);
    float bvA = biasp[nA], bvB = biasp[nB];
    int   mr  = m0 + (lane >> 4) * 8;
#pragma unroll
    for (int r = 0; r < 8; ++r) {
        out[(size_t)(mr + r) * N + nA]      = acc00[r] + bvA;
        out[(size_t)(mr + 16 + r) * N + nA] = acc10[r] + bvA;
        out[(size_t)(mr + r) * N + nB]      = acc01[r] + bvB;
        out[(size_t)(mr + 16 + r) * N + nB] = acc11[r] + bvB;
    }
}

// Fused LSTM step: gates_r = h_prev @ Whh^T over a 256-wide interleaved gate
// tile (64 hidden units x 4 gates), then the cell update.
__global__ void __launch_bounds__(256)
lstm_step(const ushort_t* __restrict__ hprev,     // [32,512] bf16 (ignored s==0)
          const ushort_t* __restrict__ Whh,       // [2048,512] bf16, gate-interleaved
          const float* __restrict__ gates_in,     // [S*B,2048]
          float* __restrict__ cbuf,               // [32,512]
          ushort_t* __restrict__ hnext,           // [32,512] bf16
          ushort_t* __restrict__ hseq,            // [S*B,512] bf16
          int s) {
    __shared__ __attribute__((aligned(16))) ushort_t lds_a[32 * HID];
    __shared__ __attribute__((aligned(16))) float g_lds[32 * 256];
    int tid = threadIdx.x, lane = tid & 31, wave = tid >> 5;
    int n0 = blockIdx.x * 256;

    f32x8 acc00 = {}, acc10 = {}, acc01 = {}, acc11 = {};
    if (s > 0) {
        stage_a_panel(hprev, 0, lds_a);
        __syncthreads();
        int nA = n0 + wave * 32 + (lane & 15);
        wave_gemm_quad(lds_a, Whh, nA, nA + 16, lane, acc00, acc10, acc01, acc11);
    }
    {   // deposit recurrent-gate tile for the cross-gate exchange
        int mr = (lane >> 4) * 8;
        int nl = wave * 32 + (lane & 15);
#pragma unroll
        for (int r = 0; r < 8; ++r) {
            g_lds[(mr + r) * 256 + nl]           = acc00[r];
            g_lds[(mr + 16 + r) * 256 + nl]      = acc10[r];
            g_lds[(mr + r) * 256 + nl + 16]      = acc01[r];
            g_lds[(mr + 16 + r) * 256 + nl + 16] = acc11[r];
        }
    }
    __syncthreads();

#pragma unroll
    for (int it = 0; it < 8; ++it) {
        int p  = tid + it * 256;          // 0..2047 : (batch, unit) pairs
        int b  = p >> 6;                  // 0..31
        int jj = p & 63;                  // 0..63 hidden units in this tile
        const float4 gi = *reinterpret_cast<const float4*>(
            gates_in + ((size_t)s * BATCH + b) * GATES + n0 + jj * 4);
        const float4 gr = *reinterpret_cast<const float4*>(&g_lds[b * 256 + jj * 4]);
        float ig = fast_sigmoid(gr.x + gi.x);
        float fg = fast_sigmoid(gr.y + gi.y);
        float gg = fast_tanh(gr.z + gi.z);
        float og = fast_sigmoid(gr.w + gi.w);
        int   j  = (n0 >> 2) + jj;
        float cp = (s == 0) ? 0.0f : cbuf[b * HID + j];
        float c  = fg * cp + ig * gg;
        float h  = og * fast_tanh(c);
        cbuf[b * HID + j] = c;
        ushort_t hb = f32_to_bf16(h);
        hnext[b * HID + j] = hb;
        hseq[((size_t)s * BATCH + b) * HID + j] = hb;
    }
}

// out[s,b,:] = htop @ Wp^T + bias ; time un-reversed for the backward dir.
// Wp padded to VPAD rows (zero-filled); stores guarded at n < VOCAB.
__global__ void __launch_bounds__(256)
proj_gemm(const ushort_t* __restrict__ A, const ushort_t* __restrict__ Wp,
          const float* __restrict__ bias, float* __restrict__ out, int reverse) {
    __shared__ __attribute__((aligned(16))) ushort_t lds_a[32 * HID];
    int tid = threadIdx.x, lane = tid & 31, wave = tid >> 5;
    int m0 = blockIdx.y * 32;
    int nA = blockIdx.x * 256 + wave * 32 + (lane & 15);
    int nB = nA + 16;
    stage_a_panel(A, m0, lds_a);
    __syncthreads();
    f32x8 acc00 = {}, acc10 = {}, acc01 = {}, acc11 = {};
    wave_gemm_quad(lds_a, Wp, nA, nB, lane, acc00, acc10, acc01, acc11);
    int mr = m0 + (lane >> 4) * 8;
#pragma unroll
    for (int r = 0; r < 8; ++r) {
        int m  = mr + r;                         // lower M half
        int s  = m >> 5;                         // BATCH == 32
        int b  = m & 31;
        int so = reverse ? (S_LEN - 1 - s) : s;
        size_t row = ((size_t)so * BATCH + b) * VOCAB;
        if (nA < VOCAB) out[row + nA] = acc00[r] + bias[nA];
        if (nB < VOCAB) out[row + nB] = acc01[r] + bias[nB];
        m  = mr + 16 + r;                        // upper M half
        s  = m >> 5;
        b  = m & 31;
        so = reverse ? (S_LEN - 1 - s) : s;
        row = ((size_t)so * BATCH + b) * VOCAB;
        if (nA < VOCAB) out[row + nA] = acc10[r] + bias[nA];
        if (nB < VOCAB) out[row + nB] = acc11[r] + bias[nB];
    }
}

// ---- conversion / permutation kernels -------------------------------------

__global__ void cvt_x_kernel(const float* __restrict__ x, ushort_t* __restrict__ xf,
                             ushort_t* __restrict__ xb, int total) {
    int i = blockIdx.x * 256 + threadIdx.x;
    if (i >= total) return;
    ushort_t v = f32_to_bf16(x[i]);
    xf[i] = v;
    int row = BATCH * HID;                       // D == HID
    int s = i / row, rest = i % row;
    xb[(size_t)(S_LEN - 1 - s) * row + rest] = v;
}

// gate-major [4][512] rows -> interleaved rows p = j*4+g, bf16
__global__ void cvt_wperm_kernel(const float* __restrict__ W, ushort_t* __restrict__ out) {
    int i = blockIdx.x * 256 + threadIdx.x;
    if (i >= GATES * HID) return;
    int p = i / HID, k = i % HID;
    int g = p & 3, j = p >> 2;
    out[i] = f32_to_bf16(W[(size_t)(g * HID + j) * HID + k]);
}

__global__ void cvt_bias_kernel(const float* __restrict__ bih, const float* __restrict__ bhh,
                                float* __restrict__ out) {
    int p = blockIdx.x * 256 + threadIdx.x;
    if (p >= GATES) return;
    int g = p & 3, j = p >> 2;
    out[p] = bih[g * HID + j] + bhh[g * HID + j];
}

// f32 -> bf16 for n elements; zero-fill tail up to ntotal (projection pad rows)
__global__ void cvt_proj_bf16(const float* __restrict__ in, ushort_t* __restrict__ out,
                              int n, int ntotal) {
    int i = blockIdx.x * 256 + threadIdx.x;
    if (i >= ntotal) return;
    out[i] = (i < n) ? f32_to_bf16(in[i]) : (ushort_t)0;
}

// ---------------------------------------------------------------------------
extern "C" void kernel_launch(void* const* d_in, const int* in_sizes, int n_in,
                              void* d_out, int out_size, void* d_ws, size_t ws_size,
                              hipStream_t stream) {
    (void)in_sizes; (void)n_in; (void)out_size; (void)ws_size;
    const float* x        = (const float*)d_in[0];
    const float* Wih_f[2] = {(const float*)d_in[1], (const float*)d_in[5]};
    const float* Whh_f[2] = {(const float*)d_in[2], (const float*)d_in[6]};
    const float* bih_f[2] = {(const float*)d_in[3], (const float*)d_in[7]};
    const float* bhh_f[2] = {(const float*)d_in[4], (const float*)d_in[8]};
    const float* Wpr_f[2] = {(const float*)d_in[9], (const float*)d_in[11]};
    const float* bpr_f[2] = {(const float*)d_in[10], (const float*)d_in[12]};
    float* out = (float*)d_out;

    // workspace carve-up (deterministic, ~77 MB)
    char* ws = (char*)d_ws;
    auto carve = [&](size_t bytes) { char* p = ws; ws += (bytes + 255) & ~(size_t)255; return p; };
    ushort_t* xdir[2];
    xdir[0] = (ushort_t*)carve((size_t)MROWS * HID * 2);
    xdir[1] = (ushort_t*)carve((size_t)MROWS * HID * 2);
    ushort_t* Wih_p[2][2]; ushort_t* Whh_p[2][2]; float* bias_p[2][2];
    for (int d = 0; d < 2; ++d)
        for (int l = 0; l < 2; ++l) {
            Wih_p[d][l]  = (ushort_t*)carve((size_t)GATES * HID * 2);
            Whh_p[d][l]  = (ushort_t*)carve((size_t)GATES * HID * 2);
            bias_p[d][l] = (float*)carve((size_t)GATES * 4);
        }
    ushort_t* Wproj[2];
    Wproj[0] = (ushort_t*)carve((size_t)VPAD * HID * 2);
    Wproj[1] = (ushort_t*)carve((size_t)VPAD * HID * 2);
    float*    gates = (float*)carve((size_t)MROWS * GATES * 4);
    ushort_t* hseq  = (ushort_t*)carve((size_t)MROWS * HID * 2);
    ushort_t* hbuf[2];
    hbuf[0] = (ushort_t*)carve((size_t)BATCH * HID * 2);
    hbuf[1] = (ushort_t*)carve((size_t)BATCH * HID * 2);
    float*    cbuf = (float*)carve((size_t)BATCH * HID * 4);

    dim3 blk(256);

    // --- one-time conversions ---
    {
        int total = MROWS * HID;
        cvt_x_kernel<<<dim3((total + 255) / 256), blk, 0, stream>>>(x, xdir[0], xdir[1], total);
    }
    for (int d = 0; d < 2; ++d) {
        for (int l = 0; l < 2; ++l) {
            int wtotal = GATES * HID;
            cvt_wperm_kernel<<<dim3((wtotal + 255) / 256), blk, 0, stream>>>(
                Wih_f[d] + (size_t)l * GATES * HID, Wih_p[d][l]);
            cvt_wperm_kernel<<<dim3((wtotal + 255) / 256), blk, 0, stream>>>(
                Whh_f[d] + (size_t)l * GATES * HID, Whh_p[d][l]);
            cvt_bias_kernel<<<dim3(GATES / 256), blk, 0, stream>>>(
                bih_f[d] + (size_t)l * GATES, bhh_f[d] + (size_t)l * GATES, bias_p[d][l]);
        }
        int ptotal = VPAD * HID;
        cvt_proj_bf16<<<dim3((ptotal + 255) / 256), blk, 0, stream>>>(
            Wpr_f[d], Wproj[d], VOCAB * HID, ptotal);
    }

    // --- per-direction pipeline ---
    for (int d = 0; d < 2; ++d) {
        // layer 0 input-side gates (all timesteps in one WMMA GEMM)
        gates_in_gemm<<<dim3(GATES / 256, MROWS / 32), blk, 0, stream>>>(
            xdir[d], Wih_p[d][0], bias_p[d][0], gates, GATES);
        // layer 0 recurrence
        for (int s = 0; s < S_LEN; ++s)
            lstm_step<<<dim3(GATES / 256), blk, 0, stream>>>(
                hbuf[s & 1], Whh_p[d][0], gates, cbuf, hbuf[(s + 1) & 1], hseq, s);
        // layer 1 input-side gates from layer-0 h sequence
        gates_in_gemm<<<dim3(GATES / 256, MROWS / 32), blk, 0, stream>>>(
            hseq, Wih_p[d][1], bias_p[d][1], gates, GATES);
        // layer 1 recurrence (overwrites hseq with top-layer h, bf16)
        for (int s = 0; s < S_LEN; ++s)
            lstm_step<<<dim3(GATES / 256), blk, 0, stream>>>(
                hbuf[s & 1], Whh_p[d][1], gates, cbuf, hbuf[(s + 1) & 1], hseq, s);
        // vocab projection (+ un-reverse time for the backward direction)
        proj_gemm<<<dim3(VPAD / 256, MROWS / 32), blk, 0, stream>>>(
            hseq, Wproj[d], bpr_f[d], out + (size_t)d * MROWS * VOCAB, d);
    }
}
